// LWAttention_50723563765809
// MI455X (gfx1250) — matrix operand order repaired and checked by reading, not compiled, not run on hardware
//
#include <hip/hip_runtime.h>
#include <hip/hip_bf16.h>

typedef __bf16 bf16;
typedef __attribute__((ext_vector_type(16))) __bf16 v16bf;
typedef __attribute__((ext_vector_type(8)))  float  v8f;

#define B_SZ   4
#define N_TOK  25088
#define C_DIM  384
#define QKV_D  1152
#define HD     48
#define NH     8
#define NL     256
#define WS     98

#define ATTN_SCALE 0.14433756729740643f   // 48^-0.5

// A-fragment K index for 16-bit 16x32 A layout (contiguous: lane half selects K 0..15 / 16..31)
__device__ __forceinline__ int a_k(int v, int half) {
    return half * 16 + 2 * v;
}

// Issue one async 16-byte global->LDS copy (ASYNCcnt-tracked).
__device__ __forceinline__ void async_copy_b128(unsigned lds_addr, const void* gaddr) {
    asm volatile("global_load_async_to_lds_b128 %0, %1, off"
                 :: "v"(lds_addr), "v"((unsigned long long)(uintptr_t)gaddr)
                 : "memory");
}
__device__ __forceinline__ void wait_async() {
    asm volatile("s_wait_asynccnt 0x0" ::: "memory");
}

// ---------------------------------------------------------------------------
// Kernel 1: QKV GEMM.  X[100352,384] f32 x Wqkv[1152,384] f32 -> bf16 QKV
// scattered into window-permuted layout [b][l][comp][head][s][hd].
// Softmax scale is folded into the Q component here so the attention kernel
// can stage q/k/v with raw async copies.
// ---------------------------------------------------------------------------
__global__ __launch_bounds__(128) void qkv_gemm(const float* __restrict__ X,
                                                const float* __restrict__ W,
                                                bf16* __restrict__ QKV) {
    __shared__ bf16 sA[64][36];
    __shared__ bf16 sB[64][36];
    const int tid  = threadIdx.x;
    const int wid  = tid >> 5;
    const int lane = tid & 31;
    const int lm   = lane & 15, half = lane >> 4;
    const long m0  = (long)blockIdx.x * 64;
    const int  n0  = blockIdx.y * 64;

    const v8f vzero = {0.f, 0.f, 0.f, 0.f, 0.f, 0.f, 0.f, 0.f};
    v8f acc[4];
#pragma unroll
    for (int j = 0; j < 4; ++j) acc[j] = vzero;

    for (int k0 = 0; k0 < C_DIM; k0 += 32) {
        for (int i = tid; i < 512; i += 128) {
            const int r = i >> 3, c = (i & 7) * 4;
            const float4 xa = *(const float4*)(X + (m0 + r) * C_DIM + k0 + c);
            sA[r][c] = (bf16)xa.x; sA[r][c + 1] = (bf16)xa.y;
            sA[r][c + 2] = (bf16)xa.z; sA[r][c + 3] = (bf16)xa.w;
            const float4 xb = *(const float4*)(W + (long)(n0 + r) * C_DIM + k0 + c);
            sB[r][c] = (bf16)xb.x; sB[r][c + 1] = (bf16)xb.y;
            sB[r][c + 2] = (bf16)xb.z; sB[r][c + 3] = (bf16)xb.w;
        }
        __syncthreads();
        v16bf a;
#pragma unroll
        for (int v = 0; v < 8; ++v) {
            const int k = a_k(v, half);
            a[2 * v]     = sA[wid * 16 + lm][k];
            a[2 * v + 1] = sA[wid * 16 + lm][k + 1];
        }
#pragma unroll
        for (int j = 0; j < 4; ++j) {
            v16bf bb;
#pragma unroll
            for (int v = 0; v < 8; ++v) {
                const int k = half * 16 + 2 * v;
                bb[2 * v]     = sB[j * 16 + lm][k];
                bb[2 * v + 1] = sB[j * 16 + lm][k + 1];
            }
            acc[j] = __builtin_amdgcn_wmma_f32_16x16x32_bf16(
                false, a, false, bb, (short)0, acc[j], false, false);
        }
        __syncthreads();
    }

    // Scatter into window-permuted bf16 layout (Q scaled).
#pragma unroll
    for (int r = 0; r < 8; ++r) {
        const long m  = m0 + wid * 16 + half * 8 + r;
        const int  b  = (int)(m / N_TOK);
        const int  n  = (int)(m % N_TOK);
        const int  d  = n / (56 * 56);
        const int  rm = n % (56 * 56);
        const int  hh = rm / 56, ww = rm % 56;
        const int  l  = (d >> 1) * 64 + (hh / 7) * 8 + (ww / 7);
        const int  s  = (d & 1) * 49 + (hh % 7) * 7 + (ww % 7);
        const long base_bl = ((long)b * NL + l) * 3;
#pragma unroll
        for (int j = 0; j < 4; ++j) {
            const int dd   = n0 + j * 16 + lm;
            const int comp = dd / C_DIM;
            const int r2   = dd % C_DIM;
            const int head = r2 / HD;
            const int e    = r2 % HD;
            const long dst = (((base_bl + comp) * NH + head) * WS + s) * HD + e;
            const float ov = acc[j][r] * ((comp == 0) ? ATTN_SCALE : 1.0f);
            QKV[dst] = (bf16)ov;
        }
    }
}

// ---------------------------------------------------------------------------
// Kernel 2: windowed attention, one block per (b, window, head).
// q/k/v staged into LDS with GLOBAL_LOAD_ASYNC_TO_LDS_B128 (raw bf16 copies,
// Q pre-scaled). S held in registers; softmax via 16-lane shuffles; P reuses
// the q/k LDS region; O = P*V via WMMA; output scattered to token order.
// LDS row strides are 16B multiples for async B128 alignment.
// ---------------------------------------------------------------------------
#define SQ_STRIDE  72                       // elems (144 B)
#define SK_OFF     (112 * SQ_STRIDE * 2)    // 16128
#define SP_STRIDE  136                      // elems (272 B), aliases sQ+sK
#define SV_OFF     (2 * SK_OFF)             // 32256
#define SV_STRIDE  56                       // elems (112 B)
#define SMEM_BYTES (SV_OFF + 128 * SV_STRIDE * 2)  // 46592

__global__ __launch_bounds__(256) void win_attn(const bf16* __restrict__ QKV,
                                                bf16* __restrict__ AO) {
    const int blk  = blockIdx.x;
    const int head = blk & 7;
    const int l    = (blk >> 3) & 255;
    const int b    = blk >> 11;
    const int tid  = threadIdx.x;
    const int wid  = tid >> 5, lane = tid & 31;
    const int lm   = lane & 15, half = lane >> 4;

    __shared__ __align__(16) char smem[SMEM_BYTES];
    bf16 (*sQ)[SQ_STRIDE] = (bf16(*)[SQ_STRIDE])smem;
    bf16 (*sK)[SQ_STRIDE] = (bf16(*)[SQ_STRIDE])(smem + SK_OFF);
    bf16 (*sP)[SP_STRIDE] = (bf16(*)[SP_STRIDE])smem;   // aliases sQ+sK later
    bf16 (*sV)[SV_STRIDE] = (bf16(*)[SV_STRIDE])(smem + SV_OFF);

    // Zero all LDS (provides K/row padding with zeros).
    {
        int* z = (int*)smem;
        for (int i = tid; i < SMEM_BYTES / 4; i += 256) z[i] = 0;
    }
    __syncthreads();

    const long  base = (((long)(b * NL + l) * 3) * NH + head) * (long)(WS * HD);
    const bf16* Qg = QKV + base;
    const bf16* Kg = QKV + base + (long)NH * WS * HD;
    const bf16* Vg = QKV + base + 2L * NH * WS * HD;

    // Async-stage q/k/v: 98 rows x 6 chunks of 8 bf16 (16 B) each.
    {
        const unsigned qb = (unsigned)(uintptr_t)smem;
        const unsigned kb = qb + SK_OFF;
        const unsigned vb = qb + SV_OFF;
        for (int i = tid; i < WS * 6; i += 256) {
            const int s = i / 6, ec = i % 6;
            const int ge = s * HD + ec * 8;
            async_copy_b128(qb + s * (SQ_STRIDE * 2) + ec * 16, Qg + ge);
            async_copy_b128(kb + s * (SQ_STRIDE * 2) + ec * 16, Kg + ge);
            async_copy_b128(vb + s * (SV_STRIDE * 2) + ec * 16, Vg + ge);
        }
        wait_async();
    }
    __syncthreads();

    v8f accS[7];
    const v8f vzero = {0.f, 0.f, 0.f, 0.f, 0.f, 0.f, 0.f, 0.f};

    // ---- S = (q*scale) k^T : M=112, N=112, K=64 (padded) ----
    if (wid < 7) {
#pragma unroll
        for (int j = 0; j < 7; ++j) accS[j] = vzero;
#pragma unroll
        for (int kk = 0; kk < 64; kk += 32) {
            v16bf a;
#pragma unroll
            for (int v = 0; v < 8; ++v) {
                const int k = kk + a_k(v, half);
                a[2 * v]     = sQ[wid * 16 + lm][k];
                a[2 * v + 1] = sQ[wid * 16 + lm][k + 1];
            }
#pragma unroll
            for (int j = 0; j < 7; ++j) {
                v16bf bb;
#pragma unroll
                for (int v = 0; v < 8; ++v) {
                    const int k = kk + half * 16 + 2 * v;
                    bb[2 * v]     = sK[j * 16 + lm][k];
                    bb[2 * v + 1] = sK[j * 16 + lm][k + 1];
                }
                accS[j] = __builtin_amdgcn_wmma_f32_16x16x32_bf16(
                    false, a, false, bb, (short)0, accS[j], false, false);
            }
        }
    }
    __syncthreads();   // everyone done reading sQ/sK; sP may now alias them

    // ---- softmax over 98 valid cols, P -> LDS bf16 ----
    if (wid < 7) {
#pragma unroll
        for (int r = 0; r < 8; ++r) {
            float vals[7];
            float mx = -3.0e38f;
#pragma unroll
            for (int j = 0; j < 7; ++j) {
                const int c = j * 16 + lm;
                const float v = (c < WS) ? accS[j][r] : -3.0e38f;
                vals[j] = v;
                mx = fmaxf(mx, v);
            }
#pragma unroll
            for (int off = 1; off < 16; off <<= 1)
                mx = fmaxf(mx, __shfl_xor(mx, off, 16));
            float sum = 0.f;
#pragma unroll
            for (int j = 0; j < 7; ++j) {
                const float p = (vals[j] > -1.0e38f) ? __expf(vals[j] - mx) : 0.f;
                vals[j] = p;
                sum += p;
            }
#pragma unroll
            for (int off = 1; off < 16; off <<= 1)
                sum += __shfl_xor(sum, off, 16);
            const float inv = 1.f / sum;
            const int row = wid * 16 + half * 8 + r;
#pragma unroll
            for (int j = 0; j < 7; ++j)
                sP[row][j * 16 + lm] = (bf16)(vals[j] * inv);
        }
        // zero pad cols 112..127 of this wave's 16 rows (K padding for GEMM2)
        for (int idx = lane; idx < 256; idx += 32)
            sP[wid * 16 + (idx >> 4)][112 + (idx & 15)] = (bf16)0.f;
    }
    __syncthreads();

    // ---- O = P V : M=112, N=48, K=128 (padded) ----
    if (wid < 7) {
        v8f accO[3];
#pragma unroll
        for (int j = 0; j < 3; ++j) accO[j] = vzero;
#pragma unroll
        for (int kk = 0; kk < 128; kk += 32) {
            v16bf a;
#pragma unroll
            for (int v = 0; v < 8; ++v) {
                const int k = kk + a_k(v, half);
                a[2 * v]     = sP[wid * 16 + lm][k];
                a[2 * v + 1] = sP[wid * 16 + lm][k + 1];
            }
#pragma unroll
            for (int j = 0; j < 3; ++j) {
                v16bf bb;
#pragma unroll
                for (int v = 0; v < 8; ++v) {
                    const int k = kk + half * 16 + 2 * v;
                    bb[2 * v]     = sV[k][j * 16 + lm];
                    bb[2 * v + 1] = sV[k + 1][j * 16 + lm];
                }
                accO[j] = __builtin_amdgcn_wmma_f32_16x16x32_bf16(
                    false, a, false, bb, (short)0, accO[j], false, false);
            }
        }

        // reverse window permutation, store bf16 at token order
        const int dwin = l >> 6, rem2 = l & 63, hwin = rem2 >> 3, wwin = rem2 & 7;
#pragma unroll
        for (int r = 0; r < 8; ++r) {
            const int s = wid * 16 + half * 8 + r;
            if (s < WS) {
                const int dloc = s / 49, rm = s % 49, hloc = rm / 7, wloc = rm % 7;
                const int n = ((dwin * 2 + dloc) * 56 + (hwin * 7 + hloc)) * 56
                              + (wwin * 7 + wloc);
                const long rowbase = ((long)b * N_TOK + n) * C_DIM + head * HD;
#pragma unroll
                for (int j = 0; j < 3; ++j)
                    AO[rowbase + j * 16 + lm] = (bf16)accO[j][r];
            }
        }
    }
}

// ---------------------------------------------------------------------------
// Kernel 3: output projection. AO[100352,384] bf16 x Wproj[384,384] f32 + bias
// -> f32 out. A-tile staged with async B128 copies (already bf16).
// ---------------------------------------------------------------------------
__global__ __launch_bounds__(128) void proj_gemm(const bf16* __restrict__ A,
                                                 const float* __restrict__ W,
                                                 const float* __restrict__ bias,
                                                 float* __restrict__ OUT) {
    __shared__ __align__(16) bf16 sA[64][40];   // 80 B rows (16B multiple)
    __shared__ bf16 sB[64][40];
    const int tid  = threadIdx.x;
    const int wid  = tid >> 5;
    const int lane = tid & 31;
    const int lm   = lane & 15, half = lane >> 4;
    const long m0  = (long)blockIdx.x * 64;
    const int  n0  = blockIdx.y * 64;
    const unsigned sa_base = (unsigned)(uintptr_t)&sA[0][0];

    const v8f vzero = {0.f, 0.f, 0.f, 0.f, 0.f, 0.f, 0.f, 0.f};
    v8f acc[4];
#pragma unroll
    for (int j = 0; j < 4; ++j) acc[j] = vzero;

    for (int k0 = 0; k0 < C_DIM; k0 += 32) {
        // A tile: 64 rows x 4 chunks of 8 bf16 -> async copies
        for (int i = tid; i < 256; i += 128) {
            const int r = i >> 2, c = i & 3;
            async_copy_b128(sa_base + r * 80 + c * 16,
                            A + (m0 + r) * C_DIM + k0 + c * 8);
        }
        // W tile: f32 -> bf16 through VGPRs
        for (int i = tid; i < 512; i += 128) {
            const int r = i >> 3, c = (i & 7) * 4;
            const float4 xb = *(const float4*)(W + (long)(n0 + r) * C_DIM + k0 + c);
            sB[r][c] = (bf16)xb.x; sB[r][c + 1] = (bf16)xb.y;
            sB[r][c + 2] = (bf16)xb.z; sB[r][c + 3] = (bf16)xb.w;
        }
        wait_async();
        __syncthreads();
        v16bf a;
#pragma unroll
        for (int v = 0; v < 8; ++v) {
            const int k = a_k(v, half);
            a[2 * v]     = sA[wid * 16 + lm][k];
            a[2 * v + 1] = sA[wid * 16 + lm][k + 1];
        }
#pragma unroll
        for (int j = 0; j < 4; ++j) {
            v16bf bb;
#pragma unroll
            for (int v = 0; v < 8; ++v) {
                const int k = half * 16 + 2 * v;
                bb[2 * v]     = sB[j * 16 + lm][k];
                bb[2 * v + 1] = sB[j * 16 + lm][k + 1];
            }
            acc[j] = __builtin_amdgcn_wmma_f32_16x16x32_bf16(
                false, a, false, bb, (short)0, acc[j], false, false);
        }
        __syncthreads();
    }

#pragma unroll
    for (int r = 0; r < 8; ++r) {
        const long m = m0 + wid * 16 + half * 8 + r;
#pragma unroll
        for (int j = 0; j < 4; ++j) {
            const int n = n0 + j * 16 + lm;
            OUT[m * C_DIM + n] = acc[j][r] + bias[n];
        }
    }
}

// ---------------------------------------------------------------------------
extern "C" void kernel_launch(void* const* d_in, const int* in_sizes, int n_in,
                              void* d_out, int out_size, void* d_ws, size_t ws_size,
                              hipStream_t stream) {
    const float* x     = (const float*)d_in[0];
    const float* wqkv  = (const float*)d_in[1];
    const float* wproj = (const float*)d_in[2];
    const float* bproj = (const float*)d_in[3];
    float* out = (float*)d_out;

    // Workspace: bf16 QKV (window-permuted, Q pre-scaled) + bf16 attn output
    bf16* qkv = (bf16*)d_ws;
    const size_t qkv_elems = (size_t)B_SZ * NL * 3 * NH * WS * HD; // 115,605,504
    bf16* ao = qkv + qkv_elems;                                    // + 38,535,168

    dim3 g1((B_SZ * N_TOK) / 64, QKV_D / 64);   // 1568 x 18
    qkv_gemm<<<g1, 128, 0, stream>>>(x, wqkv, qkv);

    win_attn<<<dim3(B_SZ * NL * NH), 256, 0, stream>>>(qkv, ao);  // 8192 blocks

    dim3 g3((B_SZ * N_TOK) / 64, C_DIM / 64);   // 1568 x 6
    proj_gemm<<<g3, 128, 0, stream>>>(ao, wproj, bproj, out);
}
